// PEER_70007966925037
// MI455X (gfx1250) — compile-verified, matrix operand order checked by reading, not run on hardware
//
#include <hip/hip_runtime.h>
#include <hip/hip_bf16.h>
#include <math.h>

typedef __attribute__((ext_vector_type(16))) _Float16 v16h;
typedef __attribute__((ext_vector_type(8)))  float    v8f;

#define DIM      1024
#define HEADS    8
#define NUM_KEYS 256
#define TOPKV    16
#define KPH      16
#define DIM_KEY  512
#define BTOK     8192
#define NCOMB    4096   // 2*HEADS*NUM_KEYS  (fused sim width)

// ---------------------------------------------------------------------------
// RMSNorm: xn = x / max(||x||,1e-12) * sqrt(DIM) * gamma ; emit f32 + f16
// ---------------------------------------------------------------------------
__global__ void peer_rmsnorm(const float* __restrict__ x,
                             const float* __restrict__ gamma,
                             float* __restrict__ xn,
                             _Float16* __restrict__ xnh) {
  int b = blockIdx.x;
  int tid = threadIdx.x;
  const float* row = x + (size_t)b * DIM;
  float v[4];
  float p = 0.f;
#pragma unroll
  for (int i = 0; i < 4; i++) { v[i] = row[tid * 4 + i]; p += v[i] * v[i]; }
#pragma unroll
  for (int off = 16; off > 0; off >>= 1) p += __shfl_xor(p, off, 32);
  __shared__ float s[8];
  int wid = tid >> 5, lane = tid & 31;
  if (lane == 0) s[wid] = p;
  __syncthreads();
  if (tid == 0) {
    float tot = 0.f;
    for (int i = 0; i < 8; i++) tot += s[i];
    s[0] = tot;
  }
  __syncthreads();
  float norm = fmaxf(sqrtf(s[0]), 1e-12f);
  float scale = 32.0f / norm;   // sqrt(1024)=32
#pragma unroll
  for (int i = 0; i < 4; i++) {
    int d = tid * 4 + i;
    float o = v[i] * scale * gamma[d];
    xn[(size_t)b * DIM + d] = o;
    xnh[(size_t)b * DIM + d] = (_Float16)o;
  }
}

// ---------------------------------------------------------------------------
// f32 -> f16 conversions
// ---------------------------------------------------------------------------
__global__ void peer_cvt_wq(const float* __restrict__ wq, _Float16* __restrict__ wqh, int n) {
  int i = blockIdx.x * blockDim.x + threadIdx.x;
  if (i < n) wqh[i] = (_Float16)wq[i];
}

// keys in: [h][key][p][d] ; out rows: [(p*8+h)*256+key][d]  (n-major, d-contig)
__global__ void peer_cvt_keys(const float* __restrict__ keys, _Float16* __restrict__ kh) {
  int i = blockIdx.x * blockDim.x + threadIdx.x;  // 8*256*2*512 total
  int d = i & 511;
  int t = i >> 9;
  int p = t & 1;
  t >>= 1;
  int key = t & 255;
  int h = t >> 8;
  kh[((size_t)((p * 8 + h) * 256 + key)) * 512 + d] = (_Float16)keys[i];
}

// ---------------------------------------------------------------------------
// WMMA fragment loader: 16-bit 16x32 A (or 32x16 B column-striped) per ISA
// 7.12.2 layout. Lane L: g=L>>4, r=L&15 selects the row (A: M, B: N).
// Element e of the v16h holds K = ((e&8)<<1) + (e&7) + 8*g, relative to k0.
// Lowers to two global_load_b128 per fragment (verified in disasm).
// ---------------------------------------------------------------------------
__device__ inline v16h peer_load_frag(const _Float16* __restrict__ base,
                                      int ld, int r, int k0, int g) {
  const _Float16* p = base + (size_t)r * ld + k0 + g * 8;
  v16h f;
#pragma unroll
  for (int e = 0; e < 16; e++) {
    int k = ((e & 8) << 1) + (e & 7);
    f[e] = p[k];
  }
  return f;
}

// ---------------------------------------------------------------------------
// GEMM0: W2T[(p,h,key)][c] = sum_d wq[c][p*4096+h*512+d] * keys[h,key,p,d]
// Per (p,h): M=1024(c) x N=256(key) x K=512(d).
// 64x64 macro-tile per wave: 16 tiles/ph-pair of (16x16), 4 A + 4 B frags
// feed 16 WMMA per k-step.  1024 wave-tiles -> 128 blocks of 8 waves.
// ---------------------------------------------------------------------------
__global__ void __launch_bounds__(256)
peer_gemm_w2(const _Float16* __restrict__ wqh,
             const _Float16* __restrict__ kh,
             _Float16* __restrict__ w2t) {
  int wid = threadIdx.x >> 5, lane = threadIdx.x & 31;
  int tile = blockIdx.x * 8 + wid;     // [0,1024)
  int ph = tile >> 6;                  // [0,16): 64 macro-tiles per (p,h)
  int tt = tile & 63;
  int tn = tt & 3, tm = tt >> 2;       // N: 4 x 64, M: 16 x 64
  int p = ph >> 3, h = ph & 7;
  int g = lane >> 4, mn = lane & 15;
  int row0 = tm * 64, col0 = tn * 64;
  const _Float16* Abase = wqh + p * 4096 + h * 512;         // row stride 8192
  const _Float16* Bbase = kh + (size_t)ph * 256 * 512;      // row stride 512
  v8f acc[4][4] = {};
  for (int k0 = 0; k0 < 512; k0 += 32) {
    v16h a[4], bb[4];
#pragma unroll
    for (int i = 0; i < 4; i++)
      a[i] = peer_load_frag(Abase, 8192, row0 + i * 16 + mn, k0, g);
#pragma unroll
    for (int j = 0; j < 4; j++)
      bb[j] = peer_load_frag(Bbase, 512, col0 + j * 16 + mn, k0, g);
#pragma unroll
    for (int i = 0; i < 4; i++)
#pragma unroll
      for (int j = 0; j < 4; j++)
        acc[i][j] = __builtin_amdgcn_wmma_f32_16x16x32_f16(
            false, a[i], false, bb[j], (short)0, acc[i][j], false, false);
  }
  // C[M][N] -> W2T[N][M]  (n-major / k-contiguous for the sim GEMM)
#pragma unroll
  for (int j = 0; j < 4; j++) {
    int ng = ph * 256 + col0 + j * 16 + mn;
#pragma unroll
    for (int i = 0; i < 4; i++)
#pragma unroll
      for (int r = 0; r < 8; r++)
        w2t[(size_t)ng * 1024 + row0 + i * 16 + r + 8 * g] =
            (_Float16)acc[i][j][r];
  }
}

// ---------------------------------------------------------------------------
// GEMM1: sim[b][n] = sum_c xn[b][c] * W2T[n][c] ; M=8192, N=4096, K=1024.
// 64x64 macro-tile per wave (4x4 grid of 16x16): 8 frag loads -> 16 WMMA
// per k-step (~32 flop/byte from L2-resident operands).
// 128x64 = 8192 wave-tiles -> 1024 blocks of 8 waves.
// ---------------------------------------------------------------------------
__global__ void __launch_bounds__(256)
peer_gemm_sim(const _Float16* __restrict__ xnh,
              const _Float16* __restrict__ w2t,
              float* __restrict__ sim) {
  int wid = threadIdx.x >> 5, lane = threadIdx.x & 31;
  int tile = blockIdx.x * 8 + wid;     // [0,8192)
  int tn = tile & 63, tm = tile >> 6;  // N: 64 x 64, M: 128 x 64
  int g = lane >> 4, mn = lane & 15;
  int row0 = tm * 64, col0 = tn * 64;
  v8f acc[4][4] = {};
  for (int k0 = 0; k0 < 1024; k0 += 32) {
    v16h a[4], bb[4];
#pragma unroll
    for (int i = 0; i < 4; i++)
      a[i] = peer_load_frag(xnh, 1024, row0 + i * 16 + mn, k0, g);
#pragma unroll
    for (int j = 0; j < 4; j++)
      bb[j] = peer_load_frag(w2t, 1024, col0 + j * 16 + mn, k0, g);
#pragma unroll
    for (int i = 0; i < 4; i++)
#pragma unroll
      for (int j = 0; j < 4; j++)
        acc[i][j] = __builtin_amdgcn_wmma_f32_16x16x32_f16(
            false, a[i], false, bb[j], (short)0, acc[i][j], false, false);
  }
#pragma unroll
  for (int i = 0; i < 4; i++)
#pragma unroll
    for (int r = 0; r < 8; r++) {
      size_t rowoff = (size_t)(row0 + i * 16 + r + 8 * g) * NCOMB;
#pragma unroll
      for (int j = 0; j < 4; j++)
        sim[rowoff + col0 + j * 16 + mn] = acc[i][j][r];
    }
}

// ---------------------------------------------------------------------------
// Top-16 of 256 per (p,h,b), one wave per unit, iterative wave argmax.
// sc layout: sc[((p*8+h)*BTOK + b)*16 + n], sorted descending.
// ---------------------------------------------------------------------------
__global__ void peer_topk16(const float* __restrict__ sim, float* __restrict__ sc) {
  int wid = threadIdx.x >> 5, lane = threadIdx.x & 31;
  int u = blockIdx.x * 8 + wid;       // [0, 131072)
  int ph = u >> 13;
  int b = u & 8191;
  const float* row = sim + (size_t)b * NCOMB + ph * 256;
  float v[8];
#pragma unroll
  for (int s = 0; s < 8; s++) v[s] = row[s * 32 + lane];
  float* outp = sc + (size_t)u * 16;
  for (int n = 0; n < 16; n++) {
    float bv = -INFINITY; int bi = 0;
#pragma unroll
    for (int s = 0; s < 8; s++)
      if (v[s] > bv) { bv = v[s]; bi = s * 32 + lane; }
#pragma unroll
    for (int off = 16; off > 0; off >>= 1) {
      float ov = __shfl_xor(bv, off, 32);
      int oi = __shfl_xor(bi, off, 32);
      if (ov > bv || (ov == bv && oi < bi)) { bv = ov; bi = oi; }
    }
    if (lane == 0) outp[n] = bv;
    if ((bi & 31) == lane) v[bi >> 5] = -INFINITY;
  }
}

// ---------------------------------------------------------------------------
// Final: per token block (8 waves = 8 heads). Combine 16x16 outer-sum,
// top-16 positions, softmax, gather down/up (rows < 256 -> L2-resident),
// exact GELU, LDS-atomic accumulate across heads.
// ---------------------------------------------------------------------------
__global__ void peer_out(const float* __restrict__ xn, const float* __restrict__ sc,
                         const float* __restrict__ down, const float* __restrict__ up,
                         float* __restrict__ out) {
  __shared__ float s_xn[DIM];
  __shared__ float s_out[DIM];
  int b = blockIdx.x;
  int tid = threadIdx.x;
  for (int d = tid; d < DIM; d += 256) {
    s_xn[d] = xn[(size_t)b * DIM + d];
    s_out[d] = 0.f;
  }
  __syncthreads();
  int h = tid >> 5;
  int lane = tid & 31;
  const float* sx = sc + ((size_t)(h) * BTOK + b) * 16;        // p=0 side
  const float* sy = sc + ((size_t)(8 + h) * BTOK + b) * 16;    // p=1 side
  float cv[8];
#pragma unroll
  for (int s = 0; s < 8; s++) {
    int t = s * 32 + lane;
    cv[s] = sx[t >> 4] + sy[t & 15];
  }
  float selv = -INFINITY; int selt = 0;
  for (int n = 0; n < 16; n++) {
    float bv = -INFINITY; int bi = 0;
#pragma unroll
    for (int s = 0; s < 8; s++)
      if (cv[s] > bv) { bv = cv[s]; bi = s * 32 + lane; }
#pragma unroll
    for (int off = 16; off > 0; off >>= 1) {
      float ov = __shfl_xor(bv, off, 32);
      int oi = __shfl_xor(bi, off, 32);
      if (ov > bv || (ov == bv && oi < bi)) { bv = ov; bi = oi; }
    }
    if (lane == n) { selv = bv; selt = bi; }
    if ((bi & 31) == lane) cv[bi >> 5] = -INFINITY;
  }
  // softmax over the 16 selected scores (held by lanes 0..15)
  float mv = (lane < 16) ? selv : -INFINITY;
#pragma unroll
  for (int off = 16; off > 0; off >>= 1) mv = fmaxf(mv, __shfl_xor(mv, off, 32));
  float ev = (lane < 16) ? expf(selv - mv) : 0.f;
  float es = ev;
#pragma unroll
  for (int off = 16; off > 0; off >>= 1) es += __shfl_xor(es, off, 32);
  float wgt = ev / es;

  float acc[32];
#pragma unroll
  for (int i = 0; i < 32; i++) acc[i] = 0.f;
  for (int k = 0; k < 16; k++) {
    float gk = __shfl(wgt, k, 32);
    int e = __shfl(selt, k, 32);     // pk index in [0,256): reference gathers by position
    const float* dw = down + (size_t)e * DIM;
    const float* uw = up + (size_t)e * DIM;
    __builtin_prefetch(uw + lane, 0, 1);            // global_prefetch_b8: warm up row
    float part = 0.f;
#pragma unroll
    for (int i = 0; i < 32; i++) part += s_xn[lane + i * 32] * dw[lane + i * 32];
#pragma unroll
    for (int off = 16; off > 0; off >>= 1) part += __shfl_xor(part, off, 32);
    float act = 0.5f * part * (1.0f + erff(part * 0.70710678118654752f));  // exact GELU
    float gg = act * gk;
#pragma unroll
    for (int i = 0; i < 32; i++) acc[i] += gg * uw[lane + i * 32];
  }
#pragma unroll
  for (int i = 0; i < 32; i++) atomicAdd(&s_out[lane + i * 32], acc[i]);
  __syncthreads();
  for (int d = tid; d < DIM; d += 256) out[(size_t)b * DIM + d] = s_out[d];
}

__global__ void peer_copy(const float* __restrict__ src, float* __restrict__ dst, int n) {
  int i = blockIdx.x * blockDim.x + threadIdx.x;
  if (i < n) dst[i] = src[i];
}

// ---------------------------------------------------------------------------
extern "C" void kernel_launch(void* const* d_in, const int* in_sizes, int n_in,
                              void* d_out, int out_size, void* d_ws, size_t ws_size,
                              hipStream_t stream) {
  const float* x        = (const float*)d_in[0];
  const float* membrain = (const float*)d_in[1];
  const float* gamma    = (const float*)d_in[2];
  const float* w_q      = (const float*)d_in[3];
  const float* keys     = (const float*)d_in[4];
  const float* down     = (const float*)d_in[5];
  const float* up       = (const float*)d_in[6];
  float* out     = (float*)d_out;
  float* out_mem = out + (size_t)BTOK * DIM;

  // workspace carve (all chunks 256B-aligned sizes)
  char* w = (char*)d_ws;
  float*    xn  = (float*)w;    w += (size_t)BTOK * DIM * 4;           // 32 MB
  _Float16* xnh = (_Float16*)w; w += (size_t)BTOK * DIM * 2;           // 16 MB
  _Float16* wqh = (_Float16*)w; w += (size_t)DIM * 8192 * 2;           // 16 MB
  _Float16* kh  = (_Float16*)w; w += (size_t)2 * 8 * 256 * 512 * 2;    //  4 MB
  _Float16* w2t = (_Float16*)w; w += (size_t)NCOMB * DIM * 2;          //  8 MB
  float*    sim = (float*)w;    w += (size_t)BTOK * NCOMB * 4;         // 128 MB
  float*    sc  = (float*)w;    w += (size_t)2 * 8 * BTOK * 16 * 4;    //  8 MB

  peer_rmsnorm<<<BTOK, 256, 0, stream>>>(x, gamma, xn, xnh);
  peer_cvt_wq<<<(DIM * 8192) / 256, 256, 0, stream>>>(w_q, wqh, DIM * 8192);
  peer_cvt_keys<<<(2 * 8 * 256 * 512) / 256, 256, 0, stream>>>(keys, kh);
  peer_gemm_w2<<<128, 256, 0, stream>>>(wqh, kh, w2t);         // 1024 wave-tiles
  peer_gemm_sim<<<1024, 256, 0, stream>>>(xnh, w2t, sim);      // 8192 wave-tiles
  peer_topk16<<<16384, 256, 0, stream>>>(sim, sc);             // 131072 waves
  peer_out<<<BTOK, 256, 0, stream>>>(xn, sc, down, up, out);
  peer_copy<<<(BTOK * HEADS * KPH) / 256, 256, 0, stream>>>(membrain, out_mem,
                                                            BTOK * HEADS * KPH);
  (void)in_sizes; (void)n_in; (void)out_size; (void)ws_size;
}
